// GNN_32598801777143
// MI455X (gfx1250) — compile-verified, matrix operand order checked by reading, not run on hardware
//
#include <hip/hip_runtime.h>

// ---------------------------------------------------------------------------
// GNN (GIN with edge features) forward for MI455X / gfx1250, wave32 + WMMA.
// Storage fp32 (reference numerics for BN stats); GEMM operands converted to
// f16 in registers, f32 accumulate via v_wmma_f32_16x16x32_f16.
// Inner GEMM loop is LDS-only (no guards, no VMEM): both A tile and B panel
// are staged zero-padded per 64-wide K chunk.
// ---------------------------------------------------------------------------

typedef __attribute__((ext_vector_type(16))) _Float16 v16h;
typedef __attribute__((ext_vector_type(8)))  float    v8f;

#define GEMM_MODE_NONE       0  // C = A*B + bias
#define GEMM_MODE_LEAKY_ACC  1  // C = leaky_relu(A*B + bias, 0.2); Caux += C
#define GEMM_MODE_COPY_AUX   2  // C = A*B + bias; Caux = C

#define KC 64                   // K chunk staged in LDS per iteration

// Generic row-major GEMM: A[M,K] (lda=K), B[K,Ncols] (ldb=Ncols), C[M,ldc].
// Block = 256 threads = 8 waves; block covers 16 rows x 128 cols
// (grid.y tiles columns in 128-wide chunks). Each wave owns a 16x16 tile.
__global__ __launch_bounds__(256)
void gemm_wmma_kernel(const float* __restrict__ A, int lda,
                      const float* __restrict__ B, int ldb,
                      const float* __restrict__ bias,
                      float* __restrict__ C, int ldc,
                      float* __restrict__ Caux,
                      int M, int K, int Ncols, int mode)
{
    __shared__ float As[16][68];    // stride 68: b128-stageable, cf-free reads
    __shared__ float Bs[KC][132];   // stride 132: coalesced stage, cf-free reads
    const int tid  = threadIdx.x;
    const int lane = tid & 31;
    const int wave = tid >> 5;      // 0..7
    const int idx  = lane & 15;     // M (for A) / N (for B) index within tile
    const int half = lane >> 4;     // lane-half selects K sub-range
    const int row0 = blockIdx.x * 16;
    const int nblk = blockIdx.y * 128;
    const int n0   = nblk + wave * 16;
    const bool computeOK = (row0 < M) && (n0 < Ncols);  // uniform per wave
    const int cl = wave * 16 + idx; // column within the 128-wide B panel

    v8f acc = {};
    for (int kc = 0; kc < K; kc += KC) {
        __syncthreads();
        // ---- stage A tile (16 x KC), one float4 per thread, zero-padded
        {
            const int r  = tid >> 4;            // 0..15
            const int k4 = (tid & 15) << 2;     // 0..60
            const int gk = kc + k4;
            float4 v; v.x = v.y = v.z = v.w = 0.0f;
            if (gk < K && (row0 + r) < M)       // K always multiple of 4
                v = *(const float4*)(A + (size_t)(row0 + r) * lda + gk);
            *(float4*)&As[r][k4] = v;
        }
        // ---- stage B panel (KC x 128), coalesced float4, zero-padded
        #pragma unroll
        for (int it = 0; it < (KC * 128 / 4) / 256; ++it) {
            const int i  = it * 256 + tid;      // float4 granule index
            const int k  = i >> 5;              // 0..KC-1 (wave-uniform)
            const int c4 = (i & 31) << 2;       // 0..124
            const int gk = kc + k;
            float4 v; v.x = v.y = v.z = v.w = 0.0f;
            if (gk < K && (nblk + c4) < Ncols)
                v = *(const float4*)(B + (size_t)gk * ldb + nblk + c4);
            *(float4*)&Bs[k][c4] = v;
        }
        __syncthreads();
        if (computeOK) {
            #pragma unroll
            for (int k0 = 0; k0 < KC; k0 += 32) {
                v16h a, b;
                #pragma unroll
                for (int v = 0; v < 8; ++v) {
                    // K offsets for 16-bit A/B operands (ISA 7.12.2):
                    // VGPR v<4: K=2v+8*half ; v>=4: K=16+2(v-4)+8*half
                    const int kk = k0 + ((v & 3) << 1) + ((v >> 2) << 4) + (half << 3);
                    const float2 av = *(const float2*)&As[idx][kk];
                    a[2 * v]     = (_Float16)av.x;
                    a[2 * v + 1] = (_Float16)av.y;
                    b[2 * v]     = (_Float16)Bs[kk][cl];
                    b[2 * v + 1] = (_Float16)Bs[kk + 1][cl];
                }
                acc = __builtin_amdgcn_wmma_f32_16x16x32_f16(
                    false, a, false, b, (short)0, acc, false, false);
            }
        }
    }
    if (!computeOK) return;

    const int col = n0 + idx;
    const float bv = bias ? bias[col] : 0.0f;
    #pragma unroll
    for (int r = 0; r < 8; ++r) {
        const int row = row0 + r + half * 8;    // C/D: VGPR r -> rows r / r+8
        if (row < M) {
            float v = acc[r] + bv;
            if (mode == GEMM_MODE_LEAKY_ACC) v = (v >= 0.0f) ? v : 0.2f * v;
            const size_t o = (size_t)row * ldc + col;
            C[o] = v;
            if (mode == GEMM_MODE_LEAKY_ACC)     Caux[o] += v;
            else if (mode == GEMM_MODE_COPY_AUX) Caux[o]  = v;
        }
    }
}

// aggr[dst] += h[src] + ee[edge]   (one wave per edge row, float4 lanes)
__global__ __launch_bounds__(256)
void scatter_kernel(const float* __restrict__ h, const float* __restrict__ ee,
                    const int* __restrict__ src, const int* __restrict__ dst,
                    float* __restrict__ aggr, int E, int H)
{
    const int edge = blockIdx.x * 8 + (threadIdx.x >> 5);
    if (edge >= E) return;
    const int s = src[edge];
    const int d = dst[edge];
    const int c0 = (threadIdx.x & 31) << 2;
    for (int c = c0; c < H; c += 128) {
        const float4 hv = *(const float4*)(h  + (size_t)s    * H + c);
        const float4 ev = *(const float4*)(ee + (size_t)edge * H + c);
        float* o = aggr + (size_t)d * H + c;
        atomicAdd(o + 0, hv.x + ev.x);
        atomicAdd(o + 1, hv.y + ev.y);
        atomicAdd(o + 2, hv.z + ev.z);
        atomicAdd(o + 3, hv.w + ev.w);
    }
}

__global__ __launch_bounds__(256)
void fill_zero_kernel(float4* __restrict__ p, size_t n4)
{
    const size_t i = (size_t)blockIdx.x * 256 + threadIdx.x;
    if (i < n4) { float4 z; z.x = z.y = z.z = z.w = 0.0f; p[i] = z; }
}

// Column sums + sums-of-squares of Z[Nrows, Ncols] into stats[0..Ncols-1]
// (sum) and stats[Ncols..2*Ncols-1] (sumsq). stats must be pre-zeroed.
// grid = (Ncols/64, rowChunks), block = 256 (4 row-lanes x 64 cols).
__global__ __launch_bounds__(256)
void bn_stats_kernel(const float* __restrict__ Z, int Nrows, int Ncols,
                     float* __restrict__ stats)
{
    const int c  = blockIdx.x * 64 + (threadIdx.x & 63);
    const int rl = threadIdx.x >> 6;
    float s = 0.0f, s2 = 0.0f;
    for (int r = blockIdx.y * 4 + rl; r < Nrows; r += gridDim.y * 4) {
        const float v = Z[(size_t)r * Ncols + c];
        s += v; s2 += v * v;
    }
    __shared__ float ls[256], ls2[256];
    ls[threadIdx.x] = s; ls2[threadIdx.x] = s2;
    __syncthreads();
    if (threadIdx.x < 64) {
        s  = ls[threadIdx.x]  + ls[threadIdx.x + 64]  + ls[threadIdx.x + 128]  + ls[threadIdx.x + 192];
        s2 = ls2[threadIdx.x] + ls2[threadIdx.x + 64] + ls2[threadIdx.x + 128] + ls2[threadIdx.x + 192];
        atomicAdd(&stats[c], s);
        atomicAdd(&stats[Ncols + c], s2);
    }
}

// In-place: Z = act(gamma*(Z-m)/sqrt(v+eps)+beta). pa==nullptr -> ReLU,
// else PReLU with slope pa[0].
__global__ __launch_bounds__(256)
void bn_apply_kernel(float* __restrict__ Z, size_t n4, int Ncols,
                     const float* __restrict__ stats,
                     const float* __restrict__ gamma,
                     const float* __restrict__ beta,
                     float invN, const float* __restrict__ pa)
{
    const size_t i = (size_t)blockIdx.x * 256 + threadIdx.x;
    if (i >= n4) return;
    const float slope = pa ? pa[0] : 0.0f;
    float4 v = ((float4*)Z)[i];
    const int base = (int)((i * 4) % (size_t)Ncols);   // Ncols % 4 == 0
    float vv[4] = {v.x, v.y, v.z, v.w};
    #pragma unroll
    for (int j = 0; j < 4; ++j) {
        const int c   = base + j;
        const float m   = stats[c] * invN;
        const float var = stats[Ncols + c] * invN - m * m;
        float xn = (vv[j] - m) * rsqrtf(var + 1e-5f) * gamma[c] + beta[c];
        vv[j] = (xn >= 0.0f) ? xn : slope * xn;
    }
    v.x = vv[0]; v.y = vv[1]; v.z = vv[2]; v.w = vv[3];
    ((float4*)Z)[i] = v;
}

// Final narrow projection out[N,C] = Z[N,K] @ W[K,C] + b (C=10, VALU is fine).
__global__ __launch_bounds__(256)
void head_kernel(const float* __restrict__ Z, const float* __restrict__ W,
                 const float* __restrict__ b, float* __restrict__ out,
                 int Nrows, int K, int C)
{
    const int r = blockIdx.x * 16 + (threadIdx.x >> 4);
    const int c = threadIdx.x & 15;
    if (r >= Nrows || c >= C) return;
    float acc = b[c];
    const float* zr = Z + (size_t)r * K;
    for (int k = 0; k < K; ++k) acc += zr[k] * W[(size_t)k * C + c];
    out[(size_t)r * C + c] = acc;
}

extern "C" void kernel_launch(void* const* d_in, const int* in_sizes, int n_in,
                              void* d_out, int out_size, void* d_ws, size_t ws_size,
                              hipStream_t stream)
{
    (void)n_in; (void)out_size; (void)ws_size;
    const float* x    = (const float*)d_in[0];
    const int*   ei   = (const int*)  d_in[1];
    const float* e    = (const float*)d_in[2];
    const float* Wn   = (const float*)d_in[3];
    const float* bnb  = (const float*)d_in[4];
    const float* We   = (const float*)d_in[5];
    const float* be   = (const float*)d_in[6];
    const float* lW1  = (const float*)d_in[7];
    const float* lb1  = (const float*)d_in[8];
    const float* lg1  = (const float*)d_in[9];
    const float* lbt1 = (const float*)d_in[10];
    const float* lW2  = (const float*)d_in[11];
    const float* lb2  = (const float*)d_in[12];
    const float* Wo1  = (const float*)d_in[13];
    const float* bo1  = (const float*)d_in[14];
    const float* go   = (const float*)d_in[15];
    const float* bo   = (const float*)d_in[16];
    const float* pa   = (const float*)d_in[17];
    const float* Wo2  = (const float*)d_in[18];
    const float* bo2  = (const float*)d_in[19];
    float* out = (float*)d_out;

    const int H  = in_sizes[4];            // bn -> H
    const int ND = in_sizes[3] / H;        // Wn = ND*H
    const int N  = in_sizes[0] / ND;       // x  = N*ND
    const int ED = in_sizes[5] / H;        // We = ED*H
    const int E  = in_sizes[2] / ED;       // e  = E*ED
    const int H2 = 2 * H;
    const int L  = in_sizes[8] / H2;       // lb1 = L*2H
    const int C  = in_sizes[19];           // bo2 -> C
    const int* src = ei;
    const int* dst = ei + E;

    // Workspace carving (256B aligned)
    char* ws = (char*)d_ws;
    size_t off = 0;
    auto carve = [&](size_t nfl) -> float* {
        float* p = (float*)(ws + off);
        off += ((nfl * sizeof(float) + 255) / 256) * 256;
        return p;
    };
    float* ee    = carve((size_t)E * H);    // edge embeddings (reused all layers)
    float* h     = carve((size_t)N * H);
    float* hsum  = carve((size_t)N * H);
    float* aggr  = carve((size_t)N * H);
    float* z1    = carve((size_t)N * H2);
    float* stats = carve((size_t)2 * H2);

    const dim3 blk(256);
    const size_t nh4 = (size_t)N * H / 4;
    const size_t st4 = (size_t)2 * H2 / 4;
    const size_t z4  = (size_t)N * H2 / 4;

    // 1) Node encoder: h = x@Wn + bn ; hsum = h
    gemm_wmma_kernel<<<dim3((N + 15) / 16, (H + 127) / 128), blk, 0, stream>>>(
        x, ND, Wn, H, bnb, h, H, hsum, N, ND, H, GEMM_MODE_COPY_AUX);
    // 2) Edge encoder: ee = e@We + be (K=16, zero-padded in LDS staging)
    gemm_wmma_kernel<<<dim3((E + 15) / 16, (H + 127) / 128), blk, 0, stream>>>(
        e, ED, We, H, be, ee, H, nullptr, E, ED, H, GEMM_MODE_NONE);

    for (int l = 0; l < L; ++l) {
        // aggr = segment_sum(h[src] + ee, dst)
        fill_zero_kernel<<<(unsigned)((nh4 + 255) / 256), blk, 0, stream>>>(
            (float4*)aggr, nh4);
        scatter_kernel<<<(unsigned)((E + 7) / 8), blk, 0, stream>>>(
            h, ee, src, dst, aggr, E, H);
        // z1 = aggr @ W1[l] + b1[l]
        gemm_wmma_kernel<<<dim3((N + 15) / 16, (H2 + 127) / 128), blk, 0, stream>>>(
            aggr, H, lW1 + (size_t)l * H * H2, H2, lb1 + (size_t)l * H2,
            z1, H2, nullptr, N, H, H2, GEMM_MODE_NONE);
        // BatchNorm(z1) + ReLU (in place)
        fill_zero_kernel<<<(unsigned)((st4 + 255) / 256), blk, 0, stream>>>(
            (float4*)stats, st4);
        bn_stats_kernel<<<dim3(H2 / 64, 128), blk, 0, stream>>>(z1, N, H2, stats);
        bn_apply_kernel<<<(unsigned)((z4 + 255) / 256), blk, 0, stream>>>(
            z1, z4, H2, stats, lg1 + (size_t)l * H2, lbt1 + (size_t)l * H2,
            1.0f / (float)N, nullptr);
        // h = leaky_relu(z1 @ W2[l] + b2[l], 0.2); hsum += h   (fused epilogue)
        gemm_wmma_kernel<<<dim3((N + 15) / 16, (H + 127) / 128), blk, 0, stream>>>(
            z1, H2, lW2 + (size_t)l * H2 * H, H, lb2 + (size_t)l * H,
            h, H, hsum, N, H2, H, GEMM_MODE_LEAKY_ACC);
    }

    // Output head: z1 = hsum @ Wo1 + bo1 ; BN ; PReLU ; out = z1 @ Wo2 + bo2
    gemm_wmma_kernel<<<dim3((N + 15) / 16, (H2 + 127) / 128), blk, 0, stream>>>(
        hsum, H, Wo1, H2, bo1, z1, H2, nullptr, N, H, H2, GEMM_MODE_NONE);
    fill_zero_kernel<<<(unsigned)((st4 + 255) / 256), blk, 0, stream>>>(
        (float4*)stats, st4);
    bn_stats_kernel<<<dim3(H2 / 64, 128), blk, 0, stream>>>(z1, N, H2, stats);
    bn_apply_kernel<<<(unsigned)((z4 + 255) / 256), blk, 0, stream>>>(
        z1, z4, H2, stats, go, bo, 1.0f / (float)N, pa);
    head_kernel<<<(N + 15) / 16, blk, 0, stream>>>(z1, Wo2, bo2, out, N, H2, C);
}
